// TransformerLayer_35364760715541
// MI455X (gfx1250) — matrix operand order repaired and checked by reading, not compiled, not run on hardware
//
#include <hip/hip_runtime.h>
#include <hip/hip_bf16.h>

typedef __attribute__((ext_vector_type(16))) __bf16 v16bf;
typedef __attribute__((ext_vector_type(8)))  __bf16 v8bf;
typedef __attribute__((ext_vector_type(8)))  float  v8f;

#define NEG_MIN_F (-3.4028234663852886e38f)

static __device__ __forceinline__ __bf16 f2bf(float f) {
  union { float f; unsigned u; } a; a.f = f;
  unsigned r = a.u + 0x7fffu + ((a.u >> 16) & 1u);
  union { unsigned short s; __bf16 b; } c; c.s = (unsigned short)(r >> 16);
  return c.b;
}

// A-operand frag: lane = (row = lane&15, half = lane>>4); K elements
// {8h..8h+7, 16+8h..16+8h+7} -> two 16B loads.
static __device__ __forceinline__ v16bf load_frag_a(const __bf16* base, long ld, int lane) {
  const int row = lane & 15, half = lane >> 4;
  const __bf16* r = base + (long)row * ld + 8 * half;
  union { v16bf v; v8bf h[2]; } u;
  u.h[0] = *(const v8bf*)(r);
  u.h[1] = *(const v8bf*)(r + 16);
  return u.v;
}

// B-operand frag (B given transposed, rows = N): lane holds col n = lane&15,
// K elements {8h..8h+7, 16+8h..16+8h+7} -> two 16B loads.
static __device__ __forceinline__ v16bf load_frag_b(const __bf16* base, long ld, int lane) {
  const int n = lane & 15, half = lane >> 4;
  const __bf16* r = base + (long)n * ld + 8 * half;
  union { v16bf v; v8bf h[2]; } u;
  u.h[0] = *(const v8bf*)(r);
  u.h[1] = *(const v8bf*)(r + 16);
  return u.v;
}

#define WMMA_BF16(acc, a, b) \
  (acc) = __builtin_amdgcn_wmma_f32_16x16x32_bf16(false, (a), false, (b), (short)0, (acc), false, false)

static __device__ __forceinline__ v8f red16_max(v8f x) {
#pragma unroll
  for (int m = 1; m < 16; m <<= 1) {
#pragma unroll
    for (int c = 0; c < 8; ++c) x[c] = fmaxf(x[c], __shfl_xor(x[c], m, 32));
  }
  return x;
}
static __device__ __forceinline__ v8f red16_add(v8f x) {
#pragma unroll
  for (int m = 1; m < 16; m <<= 1) {
#pragma unroll
    for (int c = 0; c < 8; ++c) x[c] += __shfl_xor(x[c], m, 32);
  }
  return x;
}

// ---------------- LayerNorm (fp32 in -> bf16 out), D = 1024 -----------------
__global__ void __launch_bounds__(256)
layernorm_kernel(const float* __restrict__ x, const float* __restrict__ g,
                 const float* __restrict__ bb, __bf16* __restrict__ y) {
  __shared__ float red[256];
  const int t = threadIdx.x;
  const long base = (long)blockIdx.x * 1024;
  float v[4]; float s = 0.f;
#pragma unroll
  for (int i = 0; i < 4; ++i) { v[i] = x[base + t + i * 256]; s += v[i]; }
  red[t] = s; __syncthreads();
  for (int off = 128; off; off >>= 1) { if (t < off) red[t] += red[t + off]; __syncthreads(); }
  const float mu = red[0] * (1.f / 1024.f);
  __syncthreads();
  float s2 = 0.f;
#pragma unroll
  for (int i = 0; i < 4; ++i) { float d = v[i] - mu; s2 += d * d; }
  red[t] = s2; __syncthreads();
  for (int off = 128; off; off >>= 1) { if (t < off) red[t] += red[t + off]; __syncthreads(); }
  const float rs = rsqrtf(red[0] * (1.f / 1024.f) + 1e-6f);
#pragma unroll
  for (int i = 0; i < 4; ++i) {
    const int col = t + i * 256;
    y[base + col] = f2bf((v[i] - mu) * rs * g[col] + bb[col]);
  }
}

// ------------- fp32 W(KxN) -> bf16 Wt(NxK) (transpose + convert) ------------
__global__ void __launch_bounds__(256)
transpose_w_kernel(const float* __restrict__ W, __bf16* __restrict__ Wt, int K, int N) {
  const long i = (long)blockIdx.x * 256 + threadIdx.x;
  if (i >= (long)N * K) return;
  const int n = (int)(i / K), k = (int)(i % K);
  Wt[i] = f2bf(W[(long)k * N + n]);
}

// ------------------------- flat fp32 -> bf16 convert ------------------------
__global__ void __launch_bounds__(256)
convert_bf16_kernel(const float* __restrict__ src, __bf16* __restrict__ dst, long n) {
  const long i = (long)blockIdx.x * 256 + threadIdx.x;
  if (i < n) dst[i] = f2bf(src[i]);
}

// valsT[b][hv][d][s] : hv=0 -> special_value^T, hv=1..16 -> v head hv-1 ^T
__global__ void __launch_bounds__(256)
pack_valsT_kernel(const __bf16* __restrict__ vb, const float* __restrict__ sv,
                  __bf16* __restrict__ valsT) {
  const long total = (long)8 * 17 * 64 * 512;
  const long i = (long)blockIdx.x * 256 + threadIdx.x;
  if (i >= total) return;
  const int s = (int)(i % 512); long r = i / 512;
  const int d = (int)(r % 64); r /= 64;
  const int hv = (int)(r % 17); const int b = (int)(r / 17);
  __bf16 val;
  if (hv == 0) val = f2bf(sv[((long)(b * 512 + s)) * 64 + d]);
  else         val = vb[((long)(b * 512 + s)) * 1024 + (hv - 1) * 64 + d];
  valsT[i] = val;
}

// -------- generic WMMA GEMM: C = epilogue(A(MxK) * Bt(NxK)^T) ---------------
// Block tile 128x64, BK=32, 256 threads / 8 waves; each wave owns a 32x32
// output (2x2 frag grid -> 4 back-to-back WMMAs per K-step).
// Double-buffered LDS, software-pipelined global fetch, 1 barrier per K-step.
__global__ void __launch_bounds__(256)
gemm_bf16_kernel(const __bf16* __restrict__ A, long lda, long sA_,
                 const __bf16* __restrict__ Bt, long ldb, long sB_,
                 float* __restrict__ Cf, __bf16* __restrict__ Cb, long ldc, long sC_,
                 const float* __restrict__ bias,
                 const float* __restrict__ res, long ldr, long sR_,
                 int K, float scale, int act) {
  __shared__ __align__(16) __bf16 sA[2][128 * 32];
  __shared__ __align__(16) __bf16 sB[2][64 * 32];
  const int n0 = blockIdx.x * 64, m0 = blockIdx.y * 128, bz = blockIdx.z;
  A += (long)bz * sA_; Bt += (long)bz * sB_;
  const int tid = threadIdx.x, lane = tid & 31, w = tid >> 5;
  const int wm = (w >> 1) * 32;   // 0,32,64,96
  const int wn = (w & 1) * 32;    // 0 or 32
  const int lrow = tid >> 2, lchunk = (tid & 3) * 8;

  v8f acc00 = {}, acc01 = {}, acc10 = {}, acc11 = {};
  v8bf ra0, ra1, rb0;

  const __bf16* gA0 = A + (long)(m0 + lrow) * lda + lchunk;
  const __bf16* gA1 = A + (long)(m0 + lrow + 64) * lda + lchunk;
  const __bf16* gB0 = Bt + (long)(n0 + lrow) * ldb + lchunk;

  const int nk = K >> 5;
  // prologue: fetch + stage K-step 0
  ra0 = *(const v8bf*)(gA0);
  ra1 = *(const v8bf*)(gA1);
  rb0 = *(const v8bf*)(gB0);
  *(v8bf*)(sA[0] + lrow * 32 + lchunk) = ra0;
  *(v8bf*)(sA[0] + (lrow + 64) * 32 + lchunk) = ra1;
  *(v8bf*)(sB[0] + lrow * 32 + lchunk) = rb0;
  __syncthreads();

  int cur = 0;
  for (int kk = 0; kk < nk; ++kk) {
    const bool more = (kk + 1 < nk);
    if (more) {
      const int k1 = (kk + 1) * 32;
      ra0 = *(const v8bf*)(gA0 + k1);
      ra1 = *(const v8bf*)(gA1 + k1);
      rb0 = *(const v8bf*)(gB0 + k1);
    }
    {
      const __bf16* bufA = sA[cur];
      const __bf16* bufB = sB[cur];
      const v16bf a0 = load_frag_a(bufA + (wm + 0) * 32, 32, lane);
      const v16bf a1 = load_frag_a(bufA + (wm + 16) * 32, 32, lane);
      const v16bf b0 = load_frag_b(bufB + (wn + 0) * 32, 32, lane);
      const v16bf b1 = load_frag_b(bufB + (wn + 16) * 32, 32, lane);
      WMMA_BF16(acc00, a0, b0);
      WMMA_BF16(acc01, a0, b1);
      WMMA_BF16(acc10, a1, b0);
      WMMA_BF16(acc11, a1, b1);
    }
    if (more) {
      const int nxt = cur ^ 1;
      *(v8bf*)(sA[nxt] + lrow * 32 + lchunk) = ra0;
      *(v8bf*)(sA[nxt] + (lrow + 64) * 32 + lchunk) = ra1;
      *(v8bf*)(sB[nxt] + lrow * 32 + lchunk) = rb0;
      __syncthreads();
      cur = nxt;
    }
  }

  const int half = lane >> 4, nn = lane & 15;
  const int c0 = n0 + wn + nn;
  const float bia0 = bias ? bias[c0] : 0.f;
  const float bia1 = bias ? bias[c0 + 16] : 0.f;
#pragma unroll
  for (int mi = 0; mi < 2; ++mi) {
    const v8f* am0 = mi ? &acc10 : &acc00;
    const v8f* am1 = mi ? &acc11 : &acc01;
#pragma unroll
    for (int v = 0; v < 8; ++v) {
      const long row = m0 + wm + mi * 16 + v + (half ? 8 : 0);
      float x0 = ((*am0)[v] + bia0) * scale;
      float x1 = ((*am1)[v] + bia1) * scale;
      if (act == 1) {
        x0 = x0 > 0.f ? x0 : 0.2f * x0;
        x1 = x1 > 0.f ? x1 : 0.2f * x1;
      }
      if (res) {
        x0 += res[(long)bz * sR_ + row * ldr + c0];
        x1 += res[(long)bz * sR_ + row * ldr + c0 + 16];
      }
      const long o = (long)bz * sC_ + row * ldc + c0;
      if (Cb) { Cb[o] = f2bf(x0); Cb[o + 16] = f2bf(x1); }
      else    { Cf[o] = x0;       Cf[o + 16] = x1; }
    }
  }
}

// ------------- flash attention: heads 0..15, one block per (b,h,qtile) ------
// qb/kb: bf16 token-major (B*S,1024); valsT: (B,17,64,512) bf16.
__global__ void __launch_bounds__(128)
attn_kernel(const __bf16* __restrict__ qb, const __bf16* __restrict__ kb,
            const __bf16* __restrict__ valsT, const float* __restrict__ mask,
            __bf16* __restrict__ comb) {
  __shared__ __align__(16) __bf16 sK[64 * 64];
  __shared__ __align__(16) __bf16 sV[64 * 64];
  __shared__ __align__(16) __bf16 sP[4 * 16 * 64];
  const int qt = blockIdx.x, h = blockIdx.y, bz = blockIdx.z;
  const int tid = threadIdx.x, lane = tid & 31, w = tid >> 5;
  const int half = lane >> 4, nn = lane & 15;
  const int q0 = qt * 64 + w * 16;

  const __bf16* qbase = qb + ((long)(bz * 512 + q0)) * 1024 + h * 64;
  const v16bf aq0 = load_frag_a(qbase, 1024, lane);
  const v16bf aq1 = load_frag_a(qbase + 32, 1024, lane);

  v8f o0 = {}, o1 = {}, o2 = {}, o3 = {};
  v8f mrun, lrun;
#pragma unroll
  for (int c = 0; c < 8; ++c) { mrun[c] = -3.0e38f; lrun[c] = 0.f; }

  const __bf16* kbase = kb + (long)bz * 512 * 1024 + h * 64;
  const __bf16* vbase = valsT + ((long)(bz * 17 + h)) * 64 * 512;

  for (int kt = 0; kt < 8; ++kt) {
    __syncthreads();
#pragma unroll
    for (int i = 0; i < 4; ++i) {
      const int c = tid + i * 128;               // 0..511 chunks of 8 elems
      const int r = c >> 3, ch = (c & 7) * 8;
      *(v8bf*)(sK + r * 64 + ch) = *(const v8bf*)(kbase + (long)(kt * 64 + r) * 1024 + ch);
      *(v8bf*)(sV + r * 64 + ch) = *(const v8bf*)(vbase + (long)r * 512 + kt * 64 + ch);
    }
    __syncthreads();

    v8f s0 = {}, s1 = {}, s2 = {}, s3 = {};
#pragma unroll
    for (int st = 0; st < 2; ++st) {
      const v16bf a = st ? aq1 : aq0;
      const v16bf k0f = load_frag_b(sK + 0 * 16 * 64 + st * 32, 64, lane);
      const v16bf k1f = load_frag_b(sK + 1 * 16 * 64 + st * 32, 64, lane);
      const v16bf k2f = load_frag_b(sK + 2 * 16 * 64 + st * 32, 64, lane);
      const v16bf k3f = load_frag_b(sK + 3 * 16 * 64 + st * 32, 64, lane);
      WMMA_BF16(s0, a, k0f); WMMA_BF16(s1, a, k1f);
      WMMA_BF16(s2, a, k2f); WMMA_BF16(s3, a, k3f);
    }

    const float* mrow = mask + (long)bz * 512 + kt * 64;
    const float mv0 = mrow[nn], mv1 = mrow[16 + nn], mv2 = mrow[32 + nn], mv3 = mrow[48 + nn];
#pragma unroll
    for (int c = 0; c < 8; ++c) {
      s0[c] = s0[c] * mv0 + (1.f - mv0) * NEG_MIN_F;
      s1[c] = s1[c] * mv1 + (1.f - mv1) * NEG_MIN_F;
      s2[c] = s2[c] * mv2 + (1.f - mv2) * NEG_MIN_F;
      s3[c] = s3[c] * mv3 + (1.f - mv3) * NEG_MIN_F;
    }

    const v8f t0 = red16_max(s0), t1 = red16_max(s1), t2 = red16_max(s2), t3 = red16_max(s3);
    v8f mnew, alpha;
#pragma unroll
    for (int c = 0; c < 8; ++c) {
      mnew[c] = fmaxf(fmaxf(fmaxf(t0[c], t1[c]), fmaxf(t2[c], t3[c])), mrun[c]);
      alpha[c] = __expf(mrun[c] - mnew[c]);
    }
#pragma unroll
    for (int c = 0; c < 8; ++c) {
      s0[c] = __expf(s0[c] - mnew[c]);
      s1[c] = __expf(s1[c] - mnew[c]);
      s2[c] = __expf(s2[c] - mnew[c]);
      s3[c] = __expf(s3[c] - mnew[c]);
    }
    const v8f r0 = red16_add(s0), r1 = red16_add(s1), r2 = red16_add(s2), r3 = red16_add(s3);
#pragma unroll
    for (int c = 0; c < 8; ++c) {
      lrun[c] = lrun[c] * alpha[c] + r0[c] + r1[c] + r2[c] + r3[c];
      mrun[c] = mnew[c];
      o0[c] *= alpha[c]; o1[c] *= alpha[c]; o2[c] *= alpha[c]; o3[c] *= alpha[c];
    }

    // stage P (C-layout -> row-major bf16 in per-wave LDS region)
#pragma unroll
    for (int v = 0; v < 8; ++v) {
      const int row = v + (half ? 8 : 0);
      __bf16* p = sP + (w * 16 + row) * 64 + nn;
      p[0] = f2bf(s0[v]); p[16] = f2bf(s1[v]); p[32] = f2bf(s2[v]); p[48] = f2bf(s3[v]);
    }
    __syncthreads();

#pragma unroll
    for (int st = 0; st < 2; ++st) {
      const v16bf ap = load_frag_a(sP + w * 16 * 64 + st * 32, 64, lane);
      const v16bf v0f = load_frag_b(sV + 0 * 16 * 64 + st * 32, 64, lane);
      const v16bf v1f = load_frag_b(sV + 1 * 16 * 64 + st * 32, 64, lane);
      const v16bf v2f = load_frag_b(sV + 2 * 16 * 64 + st * 32, 64, lane);
      const v16bf v3f = load_frag_b(sV + 3 * 16 * 64 + st * 32, 64, lane);
      WMMA_BF16(o0, ap, v0f); WMMA_BF16(o1, ap, v1f);
      WMMA_BF16(o2, ap, v2f); WMMA_BF16(o3, ap, v3f);
    }
  }

  v8f inv;
#pragma unroll
  for (int c = 0; c < 8; ++c) inv[c] = 1.f / lrun[c];
  const long tokbase = (long)(bz * 512 + qt * 64 + w * 16);
#pragma unroll
  for (int v = 0; v < 8; ++v) {
    const int row = v + (half ? 8 : 0);
    const long ro = (tokbase + row) * 1088 + h * 64 + nn;
    comb[ro + 0]  = f2bf(o0[v] * inv[v]);
    comb[ro + 16] = f2bf(o1[v] * inv[v]);
    comb[ro + 32] = f2bf(o2[v] * inv[v]);
    comb[ro + 48] = f2bf(o3[v] * inv[v]);
  }
}

extern "C" void kernel_launch(void* const* d_in, const int* in_sizes, int n_in,
                              void* d_out, int out_size, void* d_ws, size_t ws_size,
                              hipStream_t stream) {
  const float* x    = (const float*)d_in[0];
  const float* mask = (const float*)d_in[1];
  const float* sat  = (const float*)d_in[2];
  const float* sval = (const float*)d_in[3];
  const float* Wq   = (const float*)d_in[4];  const float* bq = (const float*)d_in[5];
  const float* Wk   = (const float*)d_in[6];  const float* bk = (const float*)d_in[7];
  const float* Wv   = (const float*)d_in[8];  const float* bv = (const float*)d_in[9];
  const float* Wo   = (const float*)d_in[10]; const float* bo = (const float*)d_in[11];
  const float* ln1g = (const float*)d_in[12]; const float* ln1b = (const float*)d_in[13];
  const float* W1   = (const float*)d_in[14]; const float* b1 = (const float*)d_in[15];
  const float* W2   = (const float*)d_in[16]; const float* b2 = (const float*)d_in[17];
  const float* ln2g = (const float*)d_in[18]; const float* ln2b = (const float*)d_in[19];
  float* out = (float*)d_out;

  char* ws = (char*)d_ws;
  size_t off = 0;
  auto take = [&](size_t bytes) {
    char* p = ws + off;
    off += (bytes + 255) & ~(size_t)255;
    return p;
  };
  __bf16* yb    = (__bf16*)take(4096ull * 1024 * 2);        // LN1 out
  __bf16* qb    = (__bf16*)take(4096ull * 1024 * 2);
  __bf16* kb    = (__bf16*)take(4096ull * 1024 * 2);
  __bf16* vb    = (__bf16*)take(4096ull * 1024 * 2);
  __bf16* WqT   = (__bf16*)take(1024ull * 1024 * 2);
  __bf16* WkT   = (__bf16*)take(1024ull * 1024 * 2);
  __bf16* WvT   = (__bf16*)take(1024ull * 1024 * 2);
  __bf16* WoT   = (__bf16*)take(1024ull * 1088 * 2);
  __bf16* W1T   = (__bf16*)take(4096ull * 1024 * 2);
  __bf16* W2T   = (__bf16*)take(1024ull * 4096 * 2);
  __bf16* saB   = (__bf16*)take(8ull * 512 * 512 * 2);
  __bf16* valsT = (__bf16*)take(8ull * 17 * 64 * 512 * 2);
  __bf16* comb  = (__bf16*)take(4096ull * 1088 * 2);
  float*  x2    = (float*)take(4096ull * 1024 * 4);
  __bf16* y2b   = (__bf16*)take(4096ull * 1024 * 2);
  __bf16* hb    = (__bf16*)take(4096ull * 4096 * 2);

  // 1) LN1
  layernorm_kernel<<<4096, 256, 0, stream>>>(x, ln1g, ln1b, yb);
  // 2) weight conversions (fp32 KxN -> bf16 NxK)
  transpose_w_kernel<<<4096, 256, 0, stream>>>(Wq, WqT, 1024, 1024);
  transpose_w_kernel<<<4096, 256, 0, stream>>>(Wk, WkT, 1024, 1024);
  transpose_w_kernel<<<4096, 256, 0, stream>>>(Wv, WvT, 1024, 1024);
  transpose_w_kernel<<<4352, 256, 0, stream>>>(Wo, WoT, 1088, 1024);
  transpose_w_kernel<<<16384, 256, 0, stream>>>(W1, W1T, 1024, 4096);
  transpose_w_kernel<<<16384, 256, 0, stream>>>(W2, W2T, 4096, 1024);
  convert_bf16_kernel<<<8192, 256, 0, stream>>>(sat, saB, 8ll * 512 * 512);

  // 3) QKV projections (scale DH^-0.5 folded into Q); tiles 128x64
  gemm_bf16_kernel<<<dim3(16, 32, 1), 256, 0, stream>>>(
      yb, 1024, 0, WqT, 1024, 0, nullptr, qb, 1024, 0, bq, nullptr, 0, 0, 1024, 0.125f, 0);
  gemm_bf16_kernel<<<dim3(16, 32, 1), 256, 0, stream>>>(
      yb, 1024, 0, WkT, 1024, 0, nullptr, kb, 1024, 0, bk, nullptr, 0, 0, 1024, 1.0f, 0);
  gemm_bf16_kernel<<<dim3(16, 32, 1), 256, 0, stream>>>(
      yb, 1024, 0, WvT, 1024, 0, nullptr, vb, 1024, 0, bv, nullptr, 0, 0, 1024, 1.0f, 0);

  // 4) values^T (special_value at index 0, v heads at 1..16)
  pack_valsT_kernel<<<17408, 256, 0, stream>>>(vb, sval, valsT);

  // 5) attention heads 0..15 -> comb[:, 0:1024]
  attn_kernel<<<dim3(8, 16, 8), 128, 0, stream>>>(qb, kb, valsT, mask, comb);

  // 6) special head 16: special_attn @ v[15] -> comb[:, 1024:1088]
  gemm_bf16_kernel<<<dim3(1, 4, 8), 256, 0, stream>>>(
      saB, 512, 512ll * 512, valsT + 16ll * 64 * 512, 512, 17ll * 64 * 512,
      nullptr, comb + 16 * 64, 1088, 512ll * 1088,
      nullptr, nullptr, 0, 0, 512, 1.0f, 0);

  // 7) output projection + residual: x2 = x + comb @ Wo + bo
  gemm_bf16_kernel<<<dim3(16, 32, 1), 256, 0, stream>>>(
      comb, 1088, 0, WoT, 1088, 0, x2, nullptr, 1024, 0, bo, x, 1024, 0, 1088, 1.0f, 0);

  // 8) LN2
  layernorm_kernel<<<4096, 256, 0, stream>>>(x2, ln2g, ln2b, y2b);

  // 9) FFN1 with leaky-relu -> hb
  gemm_bf16_kernel<<<dim3(64, 32, 1), 256, 0, stream>>>(
      y2b, 1024, 0, W1T, 1024, 0, nullptr, hb, 4096, 0, b1, nullptr, 0, 0, 1024, 1.0f, 1);

  // 10) FFN2 + residual -> out
  gemm_bf16_kernel<<<dim3(16, 32, 1), 256, 0, stream>>>(
      hb, 4096, 0, W2T, 4096, 0, out, nullptr, 1024, 0, b2, x2, 1024, 0, 4096, 1.0f, 0);
}
